// BackboneCSAI_20212116095464
// MI455X (gfx1250) — matrix operand order repaired and checked by reading, not compiled, not run on hardware
//
#include <hip/hip_runtime.h>
#include <hip/hip_bf16.h>
#include <math.h>

typedef _Float16 half_t;
typedef __attribute__((ext_vector_type(16))) _Float16 v16h;
typedef __attribute__((ext_vector_type(8)))  float    v8f;

// ---- problem dims (fixed) ----
constexpr int DB   = 256;    // batch
constexpr int DT_  = 96;     // T
constexpr int DF_  = 256;    // F
constexpr int DC_  = 256;    // C
constexpr int DHH  = 512;    // H
constexpr int DFFD = 64;
constexpr int T2_  = 192;    // 2T
constexpr int BT_  = DB * DT_;        // 24576
constexpr int B2T_ = DB * T2_;        // 49152
constexpr int LDQ_ = T2_ * 3 * DC_;   // 147456, row stride of qkv (per batch row)

enum { M_BIAS = 0, M_DECAY = 1, M_ADDPE = 2, M_ADDRES = 3, M_GELU = 4, M_GAMMAH = 5 };

#define WMMA16(a, b, c) \
  __builtin_amdgcn_wmma_f32_16x16x32_f16(false, (a), false, (b), (short)0, (c), false, false)

// ---- fragment loaders ----
// A fragment (16x32 f16) from f32 row-major source; p points at row_base + lane_hi*16.
__device__ __forceinline__ v16h frag_a_f32(const float* __restrict__ p) {
  float t[16];
  *(float4*)(t + 0)  = *(const float4*)(p + 0);
  *(float4*)(t + 4)  = *(const float4*)(p + 4);
  *(float4*)(t + 8)  = *(const float4*)(p + 8);
  *(float4*)(t + 12) = *(const float4*)(p + 12);
  v16h a;
#pragma unroll
  for (int i = 0; i < 16; ++i) a[i] = (half_t)t[i];
  return a;
}
// B fragment (32x16, N x K row-major source): 16 consecutive f32 starting at p.
__device__ __forceinline__ v16h frag_b_f32(const float* __restrict__ p) {
  float t[16];
  *(float4*)(t + 0)  = *(const float4*)(p + 0);
  *(float4*)(t + 4)  = *(const float4*)(p + 4);
  *(float4*)(t + 8)  = *(const float4*)(p + 8);
  *(float4*)(t + 12) = *(const float4*)(p + 12);
  v16h b;
#pragma unroll
  for (int i = 0; i < 16; ++i) b[i] = (half_t)t[i];
  return b;
}

// ---- generic WMMA GEMM:  out = epilogue(A[MxK,f32] @ W[NxK,f16]^T + bias) ----
// block = 256 threads (8 waves), tile = 32(M) x 64(N), one 16x16 WMMA tile per wave/ntile.
__global__ __launch_bounds__(256) void k_gemm(
    const float* __restrict__ A, int lda,
    const half_t* __restrict__ W,
    const float* __restrict__ bias,
    float* __restrict__ out,
    int M, int N, int K, int mode,
    const float* __restrict__ aux,
    const float* __restrict__ pe, int peOff) {
  const int tid = threadIdx.x, w = tid >> 5, lane = tid & 31;
  const int lh16 = lane >> 4, ln15 = lane & 15;
  const int mT = w & 1, nT = w >> 1;
  const int m0 = blockIdx.y * 32 + mT * 16;
  const int n0 = blockIdx.x * 64 + nT * 16;
  v8f acc = {0.f, 0.f, 0.f, 0.f, 0.f, 0.f, 0.f, 0.f};
  const float*  ap = A + (size_t)(m0 + ln15) * lda + lh16 * 16;
  const half_t* wp = W + (size_t)(n0 + ln15) * K + lh16 * 16;
  for (int kb = 0; kb < K; kb += 32) {
    v16h aF = frag_a_f32(ap + kb);
    v16h bF = *(const v16h*)(const void*)(wp + kb);
    acc = WMMA16(aF, bF, acc);
  }
  const int colg = n0 + ln15;
  const float bv = bias ? bias[colg] : 0.f;
#pragma unroll
  for (int r = 0; r < 8; ++r) {
    const int rowg = m0 + r + 8 * lh16;
    float val = acc[r] + bv;
    const size_t oi = (size_t)rowg * N + colg;
    if (mode == M_BIAS) {
      out[oi] = val;
    } else if (mode == M_DECAY) {
      float s = aux[oi];
      float sg = (s > 0.f) ? 1.f : ((s < 0.f) ? -1.f : 0.f);
      out[oi] = 0.5f * (1.f - tanhf(sg * fabsf(val)));
    } else if (mode == M_ADDPE) {
      int bb = rowg / DT_, tt = rowg - bb * DT_;
      out[(size_t)(bb * T2_ + peOff + tt) * N + colg] = val + pe[(size_t)tt * N + colg];
    } else if (mode == M_ADDRES) {
      out[oi] = val + aux[oi];
    } else if (mode == M_GELU) {
      out[oi] = 0.5f * val * (1.f + erff(val * 0.70710678118654752f));
    } else { // M_GAMMAH: out = aux * exp(-relu(val)), in-place safe per element
      out[oi] = aux[oi] * expf(-fmaxf(val, 0.f));
    }
  }
}

// ---- fused batch-axis attention: per (l,h) a 256x256 attention with head dim 32 ----
// grid = (8 i-tiles, 1536 (l,h)); block = 256 threads = 8 waves.
__global__ __launch_bounds__(256) void k_attn(const float* __restrict__ qkv,
                                              float* __restrict__ ctx) {
  __shared__ float  Sf[32 * 256];
  __shared__ half_t Ph[32 * 256];
  const int tid = threadIdx.x, w = tid >> 5, lane = tid & 31;
  const int lh16 = lane >> 4, ln15 = lane & 15;
  const int iT = blockIdx.x, lh = blockIdx.y;
  const int l = lh >> 3, h = lh & 7;
  const float* qb = qkv + (size_t)l * 768 + h * 32;
  const int i0 = iT * 32;
  // --- S = Q K^T / sqrt(32), tile 32 x 256 into LDS ---
  const int mT = w & 1;
  v16h aF = frag_a_f32(qb + (size_t)(i0 + mT * 16 + ln15) * LDQ_ + lh16 * 16);
  const float* kbase = qb + 256;
#pragma unroll
  for (int jt = 0; jt < 4; ++jt) {
    const int jT = (w >> 1) * 4 + jt;
    v16h bF = frag_b_f32(kbase + (size_t)(jT * 16 + ln15) * LDQ_ + lh16 * 16);
    v8f c = {0.f, 0.f, 0.f, 0.f, 0.f, 0.f, 0.f, 0.f};
    c = WMMA16(aF, bF, c);
#pragma unroll
    for (int r = 0; r < 8; ++r)
      Sf[(mT * 16 + r + 8 * lh16) * 256 + jT * 16 + ln15] = c[r] * 0.17677669529663687f;
  }
  __syncthreads();
  // --- softmax over j (256), 8 threads per row ---
  {
    const int row = tid >> 3, sub = tid & 7;
    const float* sp = Sf + row * 256 + sub * 32;
    float mx = -3.0e38f;
    for (int i = 0; i < 32; ++i) mx = fmaxf(mx, sp[i]);
    mx = fmaxf(mx, __shfl_xor(mx, 1, 32));
    mx = fmaxf(mx, __shfl_xor(mx, 2, 32));
    mx = fmaxf(mx, __shfl_xor(mx, 4, 32));
    float s = 0.f;
    for (int i = 0; i < 32; ++i) s += expf(sp[i] - mx);
    s += __shfl_xor(s, 1, 32);
    s += __shfl_xor(s, 2, 32);
    s += __shfl_xor(s, 4, 32);
    const float inv = 1.f / s;
    half_t* pp = Ph + row * 256 + sub * 32;
    for (int i = 0; i < 32; ++i) pp[i] = (half_t)(expf(sp[i] - mx) * inv);
  }
  __syncthreads();
  // --- O = P V, 32 x 32 output; waves 0..3 each one 16x16 tile ---
  if (w < 4) {
    const int mT2 = w & 1, nT = w >> 1;
    const float* vb = qb + 512;
    v8f c = {0.f, 0.f, 0.f, 0.f, 0.f, 0.f, 0.f, 0.f};
    for (int kb = 0; kb < 256; kb += 32) {
      v16h a2;
      const half_t* pa = Ph + (mT2 * 16 + ln15) * 256 + kb + lh16 * 16;
#pragma unroll
      for (int i = 0; i < 8; ++i) a2[i] = pa[i];
#pragma unroll
      for (int i = 0; i < 8; ++i) a2[8 + i] = pa[8 + i];
      v16h b2;
      const int n = nT * 16 + ln15; // d index into V^T
#pragma unroll
      for (int i = 0; i < 16; ++i) {
        const int j = kb + lh16 * 16 + i;
        b2[i] = (half_t)vb[(size_t)j * LDQ_ + n];
      }
      c = WMMA16(a2, b2, c);
    }
#pragma unroll
    for (int r = 0; r < 8; ++r) {
      const int i = i0 + mT2 * 16 + r + 8 * lh16;
      const int col = h * 32 + nT * 16 + ln15;
      ctx[((size_t)i * T2_ + l) * 256 + col] = c[r];
    }
  }
}

// ---- small elementwise / reduction kernels ----
__global__ void k_cvt(const float* __restrict__ in, half_t* __restrict__ out, int n) {
  int i = blockIdx.x * 256 + threadIdx.x;
  if (i < n) out[i] = (half_t)in[i];
}
__global__ void k_cvt_offdiag(const float* __restrict__ in, half_t* __restrict__ out) {
  int i = blockIdx.x * 256 + threadIdx.x; // 65536
  int r = i >> 8, c = i & 255;
  out[i] = (r == c) ? (half_t)0.f : (half_t)in[i];
}
__global__ void k_pe(float* __restrict__ pe) {
  int i = blockIdx.x * 256 + threadIdx.x; // 24576
  int t = i >> 8, c = i & 255;
  float div = expf(-(float)((c >> 1) * 2) * (logf(10000.f) / 256.f));
  float ang = (float)t * div;
  pe[i] = (c & 1) ? cosf(ang) : sinf(ang);
}
__global__ void k_dd(const float* __restrict__ deltas, const float* __restrict__ intervals,
                     float* __restrict__ dd) {
  size_t i = (size_t)blockIdx.x * 256 + threadIdx.x; // 6291456
  dd[i] = deltas[i] - intervals[i & 255];
}
__global__ void k_ln(float* __restrict__ buf, const float* __restrict__ g,
                     const float* __restrict__ b) {
  const int row = blockIdx.x, tid = threadIdx.x;
  const int w = tid >> 5, lane = tid & 31;
  float v = buf[(size_t)row * 256 + tid];
  __shared__ float sm[8], sq[8];
  float s = v;
#pragma unroll
  for (int m = 16; m; m >>= 1) s += __shfl_xor(s, m, 32);
  if (!lane) sm[w] = s;
  __syncthreads();
  float mean = 0.f;
#pragma unroll
  for (int i = 0; i < 8; ++i) mean += sm[i];
  mean *= (1.f / 256.f);
  float d = v - mean, q = d * d;
#pragma unroll
  for (int m = 16; m; m >>= 1) q += __shfl_xor(q, m, 32);
  if (!lane) sq[w] = q;
  __syncthreads();
  float var = 0.f;
#pragma unroll
  for (int i = 0; i < 8; ++i) var += sq[i];
  var *= (1.f / 256.f);
  buf[(size_t)row * 256 + tid] = d * rsqrtf(var + 1e-5f) * g[tid] + b[tid];
}
__global__ void k_h0(const float* __restrict__ op1, const float* __restrict__ wop2,
                     const float* __restrict__ bop2, float* __restrict__ h) {
  int idx = blockIdx.x * 256 + threadIdx.x; // 131072
  int b = idx >> 9, j = idx & 511;
  const float* p = op1 + (size_t)b * T2_ * 512 + j;
  float s = 0.f;
  for (int l = 0; l < T2_; ++l) s += p[(size_t)l * 512] * wop2[l];
  h[idx] = s + bop2[0];
}
__global__ void k_zero(float* __restrict__ p) { p[threadIdx.x] = 0.f; }
__global__ void k_prestep(const float* __restrict__ x, const float* __restrict__ deltas,
                          const float* __restrict__ mask, const float* __restrict__ xh,
                          const float* __restrict__ wdx, const float* __restrict__ bdx,
                          float* __restrict__ xr, float* __restrict__ cat1, int t) {
  int idx = blockIdx.x * 256 + threadIdx.x; // 65536
  int b = idx >> 8, f = idx & 255;
  size_t si = (size_t)(b * DT_ + t) * DF_ + f;
  float m = mask[si], xv = x[si], dv = deltas[si], xhv = xh[idx];
  xr[idx] = m * xv + (1.f - m) * xhv;
  cat1[(size_t)b * 512 + f] = expf(-fmaxf(dv * wdx[f] + bdx[f], 0.f));
  cat1[(size_t)b * 512 + 256 + f] = m;
}
__global__ void k_combine(const float* __restrict__ x, const float* __restrict__ mask,
                          const float* __restrict__ beta, const float* __restrict__ xu,
                          const float* __restrict__ xh, float* __restrict__ ximp_out,
                          float* __restrict__ rec_out, float* __restrict__ cat2,
                          float* __restrict__ lossN, float* __restrict__ lossD, int t) {
  int idx = blockIdx.x * 256 + threadIdx.x; // 65536
  int b = idx >> 8, f = idx & 255;
  size_t si = (size_t)(b * DT_ + t) * DF_ + f;
  float bv = beta[idx], m = mask[si], xv = x[si];
  float xc = bv * xu[idx] + (1.f - bv) * xh[idx];
  float xi = m * xv + (1.f - m) * xc;
  ximp_out[si] = xi;
  rec_out[si] = xc;
  cat2[(size_t)b * 512 + f] = xi;
  cat2[(size_t)b * 512 + 256 + f] = m;
  float nu = fabsf(xc - xv) * m, de = m;
  const int w = threadIdx.x >> 5, lane = threadIdx.x & 31;
  __shared__ float rN[8], rD[8];
#pragma unroll
  for (int mm = 16; mm; mm >>= 1) { nu += __shfl_xor(nu, mm, 32); de += __shfl_xor(de, mm, 32); }
  if (!lane) { rN[w] = nu; rD[w] = de; }
  __syncthreads();
  if (threadIdx.x == 0) {
    float a = 0.f, d2 = 0.f;
#pragma unroll
    for (int i = 0; i < 8; ++i) { a += rN[i]; d2 += rD[i]; }
    atomicAdd(lossN + t, a);
    atomicAdd(lossD + t, d2);
  }
}
__global__ void k_gru(const float* __restrict__ gi, const float* __restrict__ gh,
                      float* __restrict__ h) {
  int idx = blockIdx.x * 256 + threadIdx.x; // 131072
  int b = idx >> 9, j = idx & 511;
  size_t gb = (size_t)b * 1536;
  float ir = gi[gb + j], iz = gi[gb + 512 + j], in_ = gi[gb + 1024 + j];
  float hr = gh[gb + j], hz = gh[gb + 512 + j], hn = gh[gb + 1024 + j];
  float r = 1.f / (1.f + expf(-(ir + hr)));
  float z = 1.f / (1.f + expf(-(iz + hz)));
  float n = tanhf(in_ + r * hn);
  h[idx] = (1.f - z) * n + z * h[idx];
}
__global__ void k_copy(const float* __restrict__ src, float* __restrict__ dst) {
  int i = blockIdx.x * 256 + threadIdx.x;
  dst[i] = src[i];
}
__global__ void k_lossfin(const float* __restrict__ lossN, const float* __restrict__ lossD,
                          float* __restrict__ out) {
  float s = 0.f;
  for (int t = 0; t < DT_; ++t) s += lossN[t] / (lossD[t] + 1e-12f);
  out[0] = s;
}

extern "C" void kernel_launch(void* const* d_in, const int* in_sizes, int n_in,
                              void* d_out, int out_size, void* d_ws, size_t ws_size,
                              hipStream_t stream) {
  (void)in_sizes; (void)n_in; (void)out_size; (void)ws_size;
  const float* x        = (const float*)d_in[0];
  const float* mask     = (const float*)d_in[1];
  const float* deltas   = (const float*)d_in[2];
  const float* last_obs = (const float*)d_in[3];
  const float* intervals= (const float*)d_in[4];
  const float* Wo  = (const float*)d_in[5];  const float* bo  = (const float*)d_in[6];
  const float* Wip = (const float*)d_in[7];  const float* bip = (const float*)d_in[8];
  const float* Wqkv= (const float*)d_in[9];  const float* bqkv= (const float*)d_in[10];
  const float* Wao = (const float*)d_in[11]; const float* bao = (const float*)d_in[12];
  const float* g1  = (const float*)d_in[13]; const float* be1 = (const float*)d_in[14];
  const float* W1  = (const float*)d_in[15]; const float* b1  = (const float*)d_in[16];
  const float* W2  = (const float*)d_in[17]; const float* b2  = (const float*)d_in[18];
  const float* g2  = (const float*)d_in[19]; const float* be2 = (const float*)d_in[20];
  const float* Wop1= (const float*)d_in[21]; const float* bop1= (const float*)d_in[22];
  const float* wop2= (const float*)d_in[23]; const float* bop2= (const float*)d_in[24];
  const float* Wdh = (const float*)d_in[25]; const float* bdh = (const float*)d_in[26];
  const float* wdx = (const float*)d_in[27]; const float* bdx = (const float*)d_in[28];
  const float* Wh  = (const float*)d_in[29]; const float* bh  = (const float*)d_in[30];
  const float* Wfr = (const float*)d_in[31]; const float* bfr = (const float*)d_in[32];
  const float* Wwc = (const float*)d_in[33]; const float* bwc = (const float*)d_in[34];
  const float* Wih = (const float*)d_in[35]; const float* bih = (const float*)d_in[36];
  const float* Whh = (const float*)d_in[37]; const float* bhh = (const float*)d_in[38];

  float* out      = (float*)d_out;
  float* out_ximp = out;
  float* out_rec  = out + (size_t)DB * DT_ * DF_;
  float* out_h    = out_rec + (size_t)DB * DT_ * DF_;
  float* out_loss = out_h + (size_t)DB * DHH;

  char* wsp = (char*)d_ws;
  size_t off = 0;
  auto alloc = [&](size_t bytes) -> void* {
    void* p = wsp + off;
    off += (bytes + 255) & ~(size_t)255;
    return p;
  };
  // f16 weights
  half_t* Wo16  = (half_t*)alloc(65536 * 2);
  half_t* Wip16 = (half_t*)alloc(65536 * 2);
  half_t* Wqkv16= (half_t*)alloc(196608 * 2);
  half_t* Wao16 = (half_t*)alloc(65536 * 2);
  half_t* W116  = (half_t*)alloc(16384 * 2);
  half_t* W216  = (half_t*)alloc(16384 * 2);
  half_t* Wop116= (half_t*)alloc(131072 * 2);
  half_t* Wdh16 = (half_t*)alloc(131072 * 2);
  half_t* Wh16  = (half_t*)alloc(131072 * 2);
  half_t* Wfr16 = (half_t*)alloc(65536 * 2);
  half_t* Wwc16 = (half_t*)alloc(131072 * 2);
  half_t* Wih16 = (half_t*)alloc(786432 * 2);
  half_t* Whh16 = (half_t*)alloc(786432 * 2);
  // f32 buffers
  float* pe    = (float*)alloc((size_t)24576 * 4);
  float* data  = (float*)alloc((size_t)B2T_ * 256 * 4);          // 12.58M
  float* qkvb  = (float*)alloc((size_t)B2T_ * 768 * 4);          // 37.75M (also dd/decay/op1)
  float* ctx   = (float*)alloc((size_t)B2T_ * 256 * 4);
  float* tmp1  = (float*)alloc((size_t)B2T_ * 256 * 4);
  float* ffb   = (float*)alloc((size_t)B2T_ * 64 * 4);
  float* hb    = (float*)alloc((size_t)DB * DHH * 4);
  float* xh    = (float*)alloc((size_t)DB * DF_ * 4);
  float* xr    = (float*)alloc((size_t)DB * DF_ * 4);
  float* cat1  = (float*)alloc((size_t)DB * 512 * 4);
  float* xu    = (float*)alloc((size_t)DB * DF_ * 4);
  float* betab = (float*)alloc((size_t)DB * DF_ * 4);
  float* cat2  = (float*)alloc((size_t)DB * 512 * 4);
  float* gib   = (float*)alloc((size_t)DB * 1536 * 4);
  float* ghb   = (float*)alloc((size_t)DB * 1536 * 4);
  float* lossN = (float*)alloc(192 * 4); // [0:96)=num, [96:192)=den
  float* lossD = lossN + 96;
  float* dd    = qkvb;                 // alias (freed before qkv written)
  float* decay = qkvb + (size_t)BT_ * DF_;
  float* op1   = qkvb;                 // alias (freed after attention)

  auto gemm = [&](const float* A, int lda, const half_t* W, const float* bias, float* o,
                  int M, int N, int K, int mode, const float* aux, const float* pebuf,
                  int peOff) {
    dim3 grid(N / 64, (M + 31) / 32);
    k_gemm<<<grid, 256, 0, stream>>>(A, lda, W, bias, o, M, N, K, mode, aux, pebuf, peOff);
  };

  // ---- weight conversion ----
  k_cvt<<<256, 256, 0, stream>>>(Wo, Wo16, 65536);
  k_cvt<<<256, 256, 0, stream>>>(Wip, Wip16, 65536);
  k_cvt<<<768, 256, 0, stream>>>(Wqkv, Wqkv16, 196608);
  k_cvt<<<256, 256, 0, stream>>>(Wao, Wao16, 65536);
  k_cvt<<<64, 256, 0, stream>>>(W1, W116, 16384);
  k_cvt<<<64, 256, 0, stream>>>(W2, W216, 16384);
  k_cvt<<<512, 256, 0, stream>>>(Wop1, Wop116, 131072);
  k_cvt<<<512, 256, 0, stream>>>(Wdh, Wdh16, 131072);
  k_cvt<<<512, 256, 0, stream>>>(Wh, Wh16, 131072);
  k_cvt_offdiag<<<256, 256, 0, stream>>>(Wfr, Wfr16);
  k_cvt<<<512, 256, 0, stream>>>(Wwc, Wwc16, 131072);
  k_cvt<<<3072, 256, 0, stream>>>(Wih, Wih16, 786432);
  k_cvt<<<3072, 256, 0, stream>>>(Whh, Whh16, 786432);

  // ---- encoder ----
  k_pe<<<96, 256, 0, stream>>>(pe);
  k_dd<<<24576, 256, 0, stream>>>(deltas, intervals, dd);
  gemm(dd, DF_, Wo16, bo, decay, BT_, DF_, DF_, M_DECAY, dd, nullptr, 0);
  gemm(last_obs, DF_, Wip16, bip, data, BT_, DC_, DF_, M_ADDPE, nullptr, pe, 0);
  gemm(decay, DF_, Wip16, bip, data, BT_, DC_, DF_, M_ADDPE, nullptr, pe, DT_);
  gemm(data, DC_, Wqkv16, bqkv, qkvb, B2T_, 768, DC_, M_BIAS, nullptr, nullptr, 0);
  k_attn<<<dim3(8, 1536), 256, 0, stream>>>(qkvb, ctx);
  gemm(ctx, DC_, Wao16, bao, tmp1, B2T_, DC_, DC_, M_ADDRES, data, nullptr, 0);
  k_ln<<<B2T_, 256, 0, stream>>>(tmp1, g1, be1);
  gemm(tmp1, DC_, W116, b1, ffb, B2T_, DFFD, DC_, M_GELU, nullptr, nullptr, 0);
  gemm(ffb, DFFD, W216, b2, data, B2T_, DC_, DFFD, M_ADDRES, tmp1, nullptr, 0);
  k_ln<<<B2T_, 256, 0, stream>>>(data, g2, be2);
  gemm(data, DC_, Wop116, bop1, op1, B2T_, DHH, DC_, M_BIAS, nullptr, nullptr, 0);
  k_h0<<<512, 256, 0, stream>>>(op1, wop2, bop2, hb);

  // ---- sequential imputation loop ----
  k_zero<<<1, 192, 0, stream>>>(lossN);
  for (int t = 0; t < DT_; ++t) {
    // h = h * exp(-relu(d_t @ Wdh^T + bdh))   (in-place, fused epilogue)
    gemm(deltas + (size_t)t * DF_, DT_ * DF_, Wdh16, bdh, hb, DB, DHH, DF_, M_GAMMAH, hb,
         nullptr, 0);
    gemm(hb, DHH, Wh16, bh, xh, DB, DF_, DHH, M_BIAS, nullptr, nullptr, 0);
    k_prestep<<<256, 256, 0, stream>>>(x, deltas, mask, xh, wdx, bdx, xr, cat1, t);
    gemm(xr, DF_, Wfr16, bfr, xu, DB, DF_, DF_, M_BIAS, nullptr, nullptr, 0);
    gemm(cat1, 512, Wwc16, bwc, betab, DB, DF_, 512, M_BIAS, nullptr, nullptr, 0);
    k_combine<<<256, 256, 0, stream>>>(x, mask, betab, xu, xh, out_ximp, out_rec, cat2,
                                       lossN, lossD, t);
    gemm(cat2, 512, Wih16, bih, gib, DB, 1536, 512, M_BIAS, nullptr, nullptr, 0);
    gemm(hb, DHH, Whh16, bhh, ghb, DB, 1536, DHH, M_BIAS, nullptr, nullptr, 0);
    k_gru<<<512, 256, 0, stream>>>(gib, ghb, hb);
  }
  k_copy<<<512, 256, 0, stream>>>(hb, out_h);
  k_lossfin<<<1, 1, 0, stream>>>(lossN, lossD, out_loss);
}
